// NovaModel_11467562680341
// MI455X (gfx1250) — matrix operand order repaired and checked by reading, not compiled, not run on hardware
//
#include <hip/hip_runtime.h>

#define DMODEL 1024
#define SEQ    2048
#define NHEAD  16
#define HDIM   64
#define DFF    4096
#define VOCAB  32000
#define NLAYER 4

typedef __bf16 bf16_t;
typedef bf16_t v16bf __attribute__((ext_vector_type(16)));
typedef float  v8f   __attribute__((ext_vector_type(8)));
typedef unsigned u32x4 __attribute__((ext_vector_type(4)));
typedef int      i32x4 __attribute__((ext_vector_type(4)));
typedef int      i32x8 __attribute__((ext_vector_type(8)));

#if defined(__has_builtin)
#if __has_builtin(__builtin_amdgcn_tensor_load_to_lds)
#define HAVE_TDM 1
#endif
#endif

union bfrag { uint4 q[2]; v16bf v; };

__device__ __forceinline__ unsigned short f2bf(float f) {
  unsigned u = __builtin_bit_cast(unsigned, f);
  unsigned r = u + 0x7FFFu + ((u >> 16) & 1u);   // round-to-nearest-even
  return (unsigned short)(r >> 16);
}

__device__ __forceinline__ v8f wmma_bf16(const bfrag& a, const bfrag& b, v8f c) {
  return __builtin_amdgcn_wmma_f32_16x16x32_bf16(false, a.v, false, b.v,
                                                 (short)0, c, false, false);
}

__device__ __forceinline__ v8f v8f_zero() {
  v8f z;
  #pragma unroll
  for (int i = 0; i < 8; ++i) z[i] = 0.0f;
  return z;
}

// ---------------------------------------------------------------------------
// TDM: DMA one 128x32 bf16 tile of A[M,K] (row-major) into LDS.
// D# per CDNA5 ISA ch.8: group0 = {flags, lds_addr, global_addr[56:0], type=2},
// group1 = {mask=0, data_size=2B, tensor_dim0=K, tensor_dim1=M,
//           tile_dim0=32, tile_dim1=128, dim0_stride=K}. 2-D -> groups 2/3 zero.
// Issued by one wave only (TDM ignores EXEC); fenced with s_wait_tensorcnt.
// ---------------------------------------------------------------------------
#ifdef HAVE_TDM
__device__ __forceinline__ void tdm_load_tile_A(unsigned short* ldst,
                                                const unsigned short* gsrc,
                                                int M, int K) {
  const unsigned lds_off = (unsigned)(size_t)(void*)ldst;   // low 32 = LDS byte addr
  const unsigned long long ga = (unsigned long long)(size_t)gsrc;
  u32x4 g0;
  g0[0] = 1u;                                              // count=1, user mode
  g0[1] = lds_off;                                         // lds_addr
  g0[2] = (unsigned)(ga & 0xFFFFFFFFull);                  // global_addr[31:0]
  g0[3] = (unsigned)((ga >> 32) & 0x01FFFFFFull)           // global_addr[56:32]
          | (2u << 30);                                    // type = 2 (image)
  i32x8 g1;
  g1[0] = (int)(1u << 16);                                 // data_size = 2 bytes
  g1[1] = (int)(((unsigned)K & 0xFFFFu) << 16);            // tensor_dim0 lo
  g1[2] = (int)(((unsigned)K >> 16) | (((unsigned)M & 0xFFFFu) << 16));
  g1[3] = (int)(((unsigned)M >> 16) | (32u << 16));        // tile_dim0 = 32
  g1[4] = 128;                                             // tile_dim1 = 128
  g1[5] = K;                                               // dim0_stride lo32
  g1[6] = 0;
  g1[7] = 0;
  const i32x4 z4 = {0, 0, 0, 0};
#if __clang_major__ >= 23
  const i32x8 z8 = {0, 0, 0, 0, 0, 0, 0, 0};
  __builtin_amdgcn_tensor_load_to_lds(g0, g1, z4, z4, z8, 0);
#else
  __builtin_amdgcn_tensor_load_to_lds(g0, g1, z4, z4, 0);
#endif
}
#endif

// ---------------------------------------------------------------------------
// x[s,:] = emb[ids[s],:] + pos[s,:]
// ---------------------------------------------------------------------------
__global__ __launch_bounds__(256)
void embed_kernel(const int* __restrict__ ids, const float* __restrict__ emb,
                  const float* __restrict__ pos, float* __restrict__ x) {
  const int s = blockIdx.x;
  const int d0 = threadIdx.x * 4;
  const int id = ids[s];
  float4 e = *(const float4*)&emb[(size_t)id * DMODEL + d0];
  float4 p = *(const float4*)&pos[(size_t)s * DMODEL + d0];
  float4 o; o.x = e.x + p.x; o.y = e.y + p.y; o.z = e.z + p.z; o.w = e.w + p.w;
  *(float4*)&x[(size_t)s * DMODEL + d0] = o;
}

// ---------------------------------------------------------------------------
// out_bf16[s,:] = bf16( x[s,:] * rsqrt(mean(x^2)+eps) * g[:] )
// ---------------------------------------------------------------------------
__global__ __launch_bounds__(256)
void rmsnorm_kernel(const float* __restrict__ x, const float* __restrict__ g,
                    unsigned short* __restrict__ out) {
  __shared__ float red[256];
  const int s = blockIdx.x, t = threadIdx.x;
  const int d0 = t * 4;
  const float* xr = x + (size_t)s * DMODEL;
  float4 xv = *(const float4*)&xr[d0];
  red[t] = xv.x * xv.x + xv.y * xv.y + xv.z * xv.z + xv.w * xv.w;
  __syncthreads();
  for (int o = 128; o > 0; o >>= 1) {
    if (t < o) red[t] += red[t + o];
    __syncthreads();
  }
  const float rms = rsqrtf(red[0] * (1.0f / DMODEL) + 1e-6f);
  unsigned short* orow = out + (size_t)s * DMODEL;
  orow[d0 + 0] = f2bf(xv.x * rms * g[d0 + 0]);
  orow[d0 + 1] = f2bf(xv.y * rms * g[d0 + 1]);
  orow[d0 + 2] = f2bf(xv.z * rms * g[d0 + 2]);
  orow[d0 + 3] = f2bf(xv.w * rms * g[d0 + 3]);
}

// ---------------------------------------------------------------------------
// t_bf16 = bf16( silu(u) * g )
// ---------------------------------------------------------------------------
__global__ __launch_bounds__(256)
void swiglu_kernel(const float* __restrict__ u, const float* __restrict__ g,
                   unsigned short* __restrict__ out) {
  const size_t i = ((size_t)blockIdx.x * 256 + threadIdx.x) * 4;
  float4 uv = *(const float4*)&u[i];
  float4 gv = *(const float4*)&g[i];
  out[i + 0] = f2bf(uv.x / (1.0f + __expf(-uv.x)) * gv.x);
  out[i + 1] = f2bf(uv.y / (1.0f + __expf(-uv.y)) * gv.y);
  out[i + 2] = f2bf(uv.z / (1.0f + __expf(-uv.z)) * gv.z);
  out[i + 3] = f2bf(uv.w / (1.0f + __expf(-uv.w)) * gv.w);
}

// ---------------------------------------------------------------------------
// GEMM: C[M,N] = A_bf16[M,K] @ W_f32[K,N]   (W -> bf16 in the tile loader)
// MODE 0: outF = C; 1: outF = resid + C; 2: outB = bf16(C*scale);
// MODE 3: outB[n*M+m] = bf16(C)  (transposed, for V^T)
// Block tile 128x128x32, 256 thr = 8 waves, wave tile 32x64 (2x4 WMMA).
// Double-buffered LDS; A tile staged by the Tensor Data Mover.
// ---------------------------------------------------------------------------
__device__ __forceinline__ void stage_A(unsigned short* dst,
                                        const unsigned short* A, long m0,
                                        int M, int K, int k0, int t, int w) {
#ifdef HAVE_TDM
  (void)t;
  if (w == 0)                                   // one TDM op per block
    tdm_load_tile_A(dst, &A[(size_t)m0 * K + k0], M, K);
#else
  (void)w; (void)M;
  #pragma unroll
  for (int i = 0; i < 2; ++i) {
    int ch = t * 2 + i;                         // 0..511 16B chunks
    int row = ch >> 2, co = (ch & 3) * 8;
    *(uint4*)&dst[row * 32 + co] =
        *(const uint4*)&A[(size_t)(m0 + row) * K + k0 + co];
  }
#endif
}

__device__ __forceinline__ void stage_B(unsigned short* dst,
                                        const float* W, long n0,
                                        int N, int k0, int t) {
  #pragma unroll
  for (int i = 0; i < 4; ++i) {
    int idx = t + i * 256;                      // 0..1023 float4s
    int kk = idx >> 5, nn = (idx & 31) * 4;
    float4 wv = *(const float4*)&W[(size_t)(k0 + kk) * N + n0 + nn];
    dst[(nn + 0) * 32 + kk] = f2bf(wv.x);
    dst[(nn + 1) * 32 + kk] = f2bf(wv.y);
    dst[(nn + 2) * 32 + kk] = f2bf(wv.z);
    dst[(nn + 3) * 32 + kk] = f2bf(wv.w);
  }
}

template <int MODE>
__global__ __launch_bounds__(256)
void gemm_bf16(const unsigned short* __restrict__ A, const float* __restrict__ W,
               float* __restrict__ outF, unsigned short* __restrict__ outB,
               const float* __restrict__ resid,
               int M, int N, int K, float scale) {
  __shared__ unsigned short lA[2][128 * 32];    // [m][k] bf16, ping-pong
  __shared__ unsigned short lB[2][128 * 32];    // [n][k] bf16 (transposed)
  const int t = threadIdx.x, lane = t & 31, w = t >> 5;
  const int wr = w & 3, wc = w >> 2;            // 4 row groups x 2 col groups
  const int mrow = lane & 15, hlf = lane >> 4;
  const long m0 = (long)blockIdx.y * 128, n0 = (long)blockIdx.x * 128;

  v8f acc[2][4];
  #pragma unroll
  for (int r = 0; r < 2; ++r)
    #pragma unroll
    for (int c = 0; c < 4; ++c) acc[r][c] = v8f_zero();

  // prologue: stage tile 0
  stage_A(&lA[0][0], A, m0, M, K, 0, t, w);
  stage_B(&lB[0][0], W, n0, N, 0, t);
  __builtin_amdgcn_s_wait_tensorcnt(0);
  __syncthreads();

  int p = 0;
  for (int k0 = 0; k0 < K; k0 += 32) {
    // stage tile k0+32 into the other buffer while computing this one
    if (k0 + 32 < K) {
      stage_A(&lA[p ^ 1][0], A, m0, M, K, k0 + 32, t, w);
      stage_B(&lB[p ^ 1][0], W, n0, N, k0 + 32, t);
    }

    // fragments (contiguous k = 16*hlf + s in A and B) + 8 WMMA
    bfrag aF[2], bF[4];
    #pragma unroll
    for (int r = 0; r < 2; ++r) {
      int base = (wr * 32 + r * 16 + mrow) * 32;
      aF[r].q[0] = *(const uint4*)&lA[p][base + 16 * hlf];
      aF[r].q[1] = *(const uint4*)&lA[p][base + 16 * hlf + 8];
    }
    #pragma unroll
    for (int c = 0; c < 4; ++c) {
      int base = (wc * 64 + c * 16 + mrow) * 32 + 16 * hlf;
      bF[c].q[0] = *(const uint4*)&lB[p][base];
      bF[c].q[1] = *(const uint4*)&lB[p][base + 8];
    }
    #pragma unroll
    for (int r = 0; r < 2; ++r)
      #pragma unroll
      for (int c = 0; c < 4; ++c)
        acc[r][c] = wmma_bf16(aF[r], bF[c], acc[r][c]);

    __builtin_amdgcn_s_wait_tensorcnt(0);       // TDM done before buffer swap
    __syncthreads();
    p ^= 1;
  }

  // epilogue: D layout -> lane holds n = mrow, rows m = v + 8*hlf
  #pragma unroll
  for (int r = 0; r < 2; ++r)
    #pragma unroll
    for (int c = 0; c < 4; ++c)
      #pragma unroll
      for (int v = 0; v < 8; ++v) {
        size_t m = (size_t)(m0 + wr * 32 + r * 16 + v + 8 * hlf);
        size_t n = (size_t)(n0 + wc * 64 + c * 16 + mrow);
        float val = acc[r][c][v];
        if (MODE == 0) outF[m * N + n] = val;
        if (MODE == 1) outF[m * N + n] = resid[m * N + n] + val;
        if (MODE == 2) outB[m * N + n] = f2bf(val * scale);
        if (MODE == 3) outB[n * (size_t)M + m] = f2bf(val);
      }
}

// ---------------------------------------------------------------------------
// Flash attention, causal. grid (SEQ/64, NHEAD), block 128 = 4 waves.
// Wave owns 16 query rows; streams 32-key chunks with online softmax.
// Q pre-scaled by 1/sqrt(HD). Inputs bf16: Q,K [S,D]; V^T [D,S]. Out bf16 [S,D].
// ---------------------------------------------------------------------------
__global__ __launch_bounds__(128)
void flash_attn_kernel(const unsigned short* __restrict__ Q,
                       const unsigned short* __restrict__ Kb,
                       const unsigned short* __restrict__ VT,
                       unsigned short* __restrict__ O) {
  __shared__ float          sc[4][16][32];   // per-wave score tiles
  __shared__ unsigned short pb[4][16][32];   // per-wave P (bf16)
  __shared__ float          sal[4][16];      // per-row alpha / 1/l
  const int t = threadIdx.x, lane = t & 31, w = t >> 5;
  const int h = blockIdx.y;
  const int q0 = blockIdx.x * 64 + w * 16;
  const int mrow = lane & 15, hlf = lane >> 4;
  const int hd0 = h * HDIM;

  bfrag aQ[2];
  const unsigned short* qrow = Q + (size_t)(q0 + mrow) * DMODEL + hd0;
  #pragma unroll
  for (int c = 0; c < 2; ++c) {
    aQ[c].q[0] = *(const uint4*)&qrow[c * 32 + 16 * hlf];
    aQ[c].q[1] = *(const uint4*)&qrow[c * 32 + 16 * hlf + 8];
  }

  v8f acc[4];
  #pragma unroll
  for (int i = 0; i < 4; ++i) acc[i] = v8f_zero();
  float mi = -1e30f, li = 0.0f;

  for (int kb = 0; kb < q0 + 16; kb += 32) {      // causal chunk range
    #pragma unroll
    for (int t2 = 0; t2 < 2; ++t2) {
      const unsigned short* krow =
          Kb + (size_t)(kb + t2 * 16 + mrow) * DMODEL + hd0;
      v8f s = v8f_zero();
      #pragma unroll
      for (int c = 0; c < 2; ++c) {
        bfrag bK;
        bK.q[0] = *(const uint4*)&krow[c * 32 + 16 * hlf];
        bK.q[1] = *(const uint4*)&krow[c * 32 + 16 * hlf + 8];
        s = wmma_bf16(aQ[c], bK, s);
      }
      const int key = kb + t2 * 16 + mrow;
      #pragma unroll
      for (int v = 0; v < 8; ++v) {
        int m = v + 8 * hlf;
        sc[w][m][t2 * 16 + mrow] = (key <= q0 + m) ? s[v] : -1e30f;
      }
    }
    asm volatile("s_wait_dscnt 0" ::: "memory");

    if (lane < 16) {                              // lane i owns row i
      const int m = lane;
      float cm = -1e30f;
      #pragma unroll
      for (int j = 0; j < 32; ++j) cm = fmaxf(cm, sc[w][m][j]);
      const float nm = fmaxf(mi, cm);
      const float al = __expf(mi - nm);
      float sum = 0.0f;
      #pragma unroll
      for (int j = 0; j < 32; ++j) {
        float pv = __expf(sc[w][m][j] - nm);
        sum += pv;
        pb[w][m][j] = f2bf(pv);
      }
      sal[w][m] = al;
      mi = nm;
      li = li * al + sum;
    }
    asm volatile("s_wait_dscnt 0" ::: "memory");

    float av[8];
    #pragma unroll
    for (int v = 0; v < 8; ++v) av[v] = sal[w][v + 8 * hlf];
    #pragma unroll
    for (int t4 = 0; t4 < 4; ++t4)
      #pragma unroll
      for (int v = 0; v < 8; ++v) acc[t4][v] *= av[v];

    bfrag aP;
    aP.q[0] = *(const uint4*)&pb[w][mrow][16 * hlf];
    aP.q[1] = *(const uint4*)&pb[w][mrow][16 * hlf + 8];
    #pragma unroll
    for (int t4 = 0; t4 < 4; ++t4) {
      const unsigned short* vrow =
          VT + (size_t)(hd0 + t4 * 16 + mrow) * SEQ + kb;
      bfrag bV;
      bV.q[0] = *(const uint4*)&vrow[16 * hlf];
      bV.q[1] = *(const uint4*)&vrow[16 * hlf + 8];
      acc[t4] = wmma_bf16(aP, bV, acc[t4]);
    }
  }

  if (lane < 16) sal[w][lane] = 1.0f / li;
  asm volatile("s_wait_dscnt 0" ::: "memory");
  float rv[8];
  #pragma unroll
  for (int v = 0; v < 8; ++v) rv[v] = sal[w][v + 8 * hlf];
  #pragma unroll
  for (int t4 = 0; t4 < 4; ++t4)
    #pragma unroll
    for (int v = 0; v < 8; ++v) {
      int m = v + 8 * hlf;
      O[(size_t)(q0 + m) * DMODEL + hd0 + t4 * 16 + mrow] =
          f2bf(acc[t4][v] * rv[v]);
    }
}

// ---------------------------------------------------------------------------
extern "C" void kernel_launch(void* const* d_in, const int* in_sizes, int n_in,
                              void* d_out, int out_size, void* d_ws,
                              size_t ws_size, hipStream_t stream) {
  (void)in_sizes; (void)n_in; (void)out_size; (void)ws_size;
  const int*   ids    = (const int*)  d_in[0];
  const float* emb    = (const float*)d_in[1];
  const float* pos    = (const float*)d_in[2];
  const float* g_attn = (const float*)d_in[3];
  const float* wq     = (const float*)d_in[4];
  const float* wk     = (const float*)d_in[5];
  const float* wv     = (const float*)d_in[6];
  const float* wo     = (const float*)d_in[7];
  const float* g_ffn  = (const float*)d_in[8];
  const float* w1     = (const float*)d_in[9];
  const float* w2     = (const float*)d_in[10];
  const float* w3     = (const float*)d_in[11];
  const float* g_fin  = (const float*)d_in[12];
  const float* w_lm   = (const float*)d_in[13];
  float* logits = (float*)d_out;

  char* ws = (char*)d_ws;
  size_t off = 0;
  auto carve = [&](size_t bytes) -> void* {
    void* p = ws + off;
    off += (bytes + 255) & ~(size_t)255;
    return p;
  };
  float*          x      = (float*)         carve((size_t)SEQ * DMODEL * 4);
  unsigned short* hbf    = (unsigned short*)carve((size_t)SEQ * DMODEL * 2);
  unsigned short* qbf    = (unsigned short*)carve((size_t)SEQ * DMODEL * 2);
  unsigned short* kbf    = (unsigned short*)carve((size_t)SEQ * DMODEL * 2);
  unsigned short* vTbf   = (unsigned short*)carve((size_t)DMODEL * SEQ * 2);
  unsigned short* attnbf = (unsigned short*)carve((size_t)SEQ * DMODEL * 2);
  float*          ubuf   = (float*)         carve((size_t)SEQ * DFF * 4);
  float*          gbuf   = (float*)         carve((size_t)SEQ * DFF * 4);
  unsigned short* tbf    = (unsigned short*)carve((size_t)SEQ * DFF * 2);

  const dim3 blk256(256), blk128(128);
  const dim3 gD(DMODEL / 128, SEQ / 128);
  const dim3 gF(DFF / 128, SEQ / 128);
  const dim3 gV(VOCAB / 128, SEQ / 128);
  const dim3 gAtt(SEQ / 64, NHEAD);
  const float qscale = 0.125f;                // 1/sqrt(HD=64)

  embed_kernel<<<SEQ, blk256, 0, stream>>>(ids, emb, pos, x);

  for (int i = 0; i < NLAYER; ++i) {
    const float* wqi = wq + (size_t)i * DMODEL * DMODEL;
    const float* wki = wk + (size_t)i * DMODEL * DMODEL;
    const float* wvi = wv + (size_t)i * DMODEL * DMODEL;
    const float* woi = wo + (size_t)i * DMODEL * DMODEL;
    const float* w1i = w1 + (size_t)i * DMODEL * DFF;
    const float* w2i = w2 + (size_t)i * DFF * DMODEL;
    const float* w3i = w3 + (size_t)i * DMODEL * DFF;

    rmsnorm_kernel<<<SEQ, blk256, 0, stream>>>(x, g_attn + (size_t)i * DMODEL, hbf);
    gemm_bf16<2><<<gD, blk256, 0, stream>>>(hbf, wqi, nullptr, qbf, nullptr,
                                            SEQ, DMODEL, DMODEL, qscale);
    gemm_bf16<2><<<gD, blk256, 0, stream>>>(hbf, wki, nullptr, kbf, nullptr,
                                            SEQ, DMODEL, DMODEL, 1.0f);
    gemm_bf16<3><<<gD, blk256, 0, stream>>>(hbf, wvi, nullptr, vTbf, nullptr,
                                            SEQ, DMODEL, DMODEL, 1.0f);
    flash_attn_kernel<<<gAtt, blk128, 0, stream>>>(qbf, kbf, vTbf, attnbf);
    gemm_bf16<1><<<gD, blk256, 0, stream>>>(attnbf, woi, x, nullptr, x,
                                            SEQ, DMODEL, DMODEL, 1.0f);

    rmsnorm_kernel<<<SEQ, blk256, 0, stream>>>(x, g_ffn + (size_t)i * DMODEL, hbf);
    gemm_bf16<0><<<gF, blk256, 0, stream>>>(hbf, w1i, ubuf, nullptr, nullptr,
                                            SEQ, DFF, DMODEL, 1.0f);
    gemm_bf16<0><<<gF, blk256, 0, stream>>>(hbf, w3i, gbuf, nullptr, nullptr,
                                            SEQ, DFF, DMODEL, 1.0f);
    swiglu_kernel<<<(SEQ * DFF) / 1024, blk256, 0, stream>>>(ubuf, gbuf, tbf);
    gemm_bf16<1><<<gD, blk256, 0, stream>>>(tbf, w2i, x, nullptr, x,
                                            SEQ, DMODEL, DFF, 1.0f);
  }

  rmsnorm_kernel<<<SEQ, blk256, 0, stream>>>(x, g_fin, hbf);
  gemm_bf16<0><<<gV, blk256, 0, stream>>>(hbf, w_lm, logits, nullptr, nullptr,
                                          SEQ, VOCAB, DMODEL, 1.0f);
}
